// Decoder_25486335935120
// MI455X (gfx1250) — matrix operand order repaired and checked by reading, not compiled, not run on hardware
//
#include <hip/hip_runtime.h>
#include <stdint.h>

#define BATCH  32
#define TSTEPS 256
#define VOCAB  8192
#define HIDDEN 1024
#define GATES  (4*HIDDEN)

typedef __attribute__((ext_vector_type(16))) __bf16 bf16x16;
typedef __attribute__((ext_vector_type(8)))  float  f32x8;

union Frag { bf16x16 v; uint4 q[2]; };

__device__ __forceinline__ unsigned short f2bf(float f) {
    unsigned u = __float_as_uint(f);
    unsigned r = ((u >> 16) & 1u) + 0x7FFFu;   // round-to-nearest-even
    return (unsigned short)((u + r) >> 16);
}
__device__ __forceinline__ float bf2f(unsigned short b) {
    return __uint_as_float(((unsigned)b) << 16);
}
__device__ __forceinline__ float sigm(float x) { return 1.0f / (1.0f + __expf(-x)); }

__device__ __forceinline__ f32x8 wmma_bf16(bf16x16 a, bf16x16 b, f32x8 c) {
    // (neg_a, A, neg_b, B, c_mod, C, reuse_a, reuse_b)
    return __builtin_amdgcn_wmma_f32_16x16x32_bf16(false, a, false, b, (short)0, c, false, false);
}

// Gather one element of packed-fragment-layout Wi: Wi[k, col], K=VOCAB, N=GATES.
__device__ __forceinline__ float wi_gather(const unsigned short* __restrict__ WiP, int k, int col) {
    int nt = col >> 4;
    int kt = k >> 5;
    int ln = (col & 15) | (((k >> 4) & 1) << 4);
    int el = k & 15;
    size_t idx = (((size_t)(nt * (VOCAB >> 5) + kt)) * 32 + ln) * 16 + el;
    return bf2f(WiP[idx]);
}

// -------------------- prep kernels --------------------

// Repack row-major fp32 W[K][N] (N = 1<<nShift) into bf16 WMMA B-fragment layout:
// tile (nt, kt) of 16 cols x 32 K; lane l<16: col=l, K=kt*32+elem; lane>=16: col=l-16, K=kt*32+16+elem.
__global__ void pack_weight_kernel(const float* __restrict__ W, unsigned short* __restrict__ out,
                                   int K, int nShift) {
    size_t idx = (size_t)blockIdx.x * 256 + threadIdx.x;
    size_t total = (size_t)K << nShift;
    if (idx >= total) return;
    int n = (int)(idx & ((1u << nShift) - 1u));
    int k = (int)(idx >> nShift);
    int nt = n >> 4, kt = k >> 5;
    int ln = (n & 15) | (((k >> 4) & 1) << 4);
    int el = k & 15;
    int ktCount = K >> 5;
    size_t pidx = (((size_t)(nt * ktCount + kt)) * 32 + ln) * 16 + el;
    out[pidx] = f2bf(W[idx]);
}

__global__ void cvt_x0_kernel(const float* __restrict__ inputs, unsigned short* __restrict__ x0) {
    int idx = blockIdx.x * 256 + threadIdx.x;        // BATCH*VOCAB
    int b = idx >> 13, v = idx & (VOCAB - 1);
    x0[idx] = f2bf(inputs[(size_t)b * (TSTEPS * VOCAB) + v]);
}

__global__ void init_c_kernel(const float* __restrict__ c0, float* __restrict__ c) {
    int idx = blockIdx.x * 256 + threadIdx.x;
    if (idx < BATCH * HIDDEN) c[idx] = c0[idx];
}

// -------------------- fused gates GEMM + LSTM pointwise --------------------
// Each WG owns 32 hidden columns [n0, n0+32) across all 4 gates.
// 8 waves x 2 WMMA tiles = 16 tiles: wave w -> gate g = w>>1, mt = w&1, nt16 in {0,1}.
// t==0: A = x0_bf16 (lda=VOCAB), B = WiP, ktCount = VOCAB/32, useGather = 0.
// t>0 : A = h_bf16  (lda=HIDDEN), B = WhP, ktCount = HIDDEN/32, useGather = 1.
__global__ void lstm_step_kernel(const unsigned short* __restrict__ A, int lda,
                                 const unsigned short* __restrict__ B, int ktCount,
                                 const unsigned short* __restrict__ WiP,
                                 const float* __restrict__ bh,
                                 const int* __restrict__ tok,
                                 float* __restrict__ c,
                                 unsigned short* __restrict__ hOut,
                                 int useGather) {
    __shared__ float zs[4][BATCH][32];   // 16 KB: z pre-activations for this WG's columns

    const int tid  = threadIdx.x;
    const int wave = tid >> 5, lane = tid & 31;
    const int g = wave >> 1, mt = wave & 1;
    const int n0 = blockIdx.x * 32;
    const int m = lane & 15, hi = lane >> 4;

    const unsigned short* Arow = A + (size_t)(mt * 16 + m) * lda + hi * 16;
    const int ntG0 = g * 64 + (n0 >> 4);            // gate block offset /16

    f32x8 acc0 = {}; f32x8 acc1 = {};
    for (int kt = 0; kt < ktCount; ++kt) {
        Frag a, b0, b1;
        a.q[0] = *(const uint4*)(Arow + kt * 32);
        a.q[1] = *(const uint4*)(Arow + kt * 32 + 8);
        const unsigned short* p0 = B + (((size_t)ntG0 * ktCount + kt) * 32 + lane) * 16;
        const unsigned short* p1 = B + (((size_t)(ntG0 + 1) * ktCount + kt) * 32 + lane) * 16;
        b0.q[0] = *(const uint4*)(p0); b0.q[1] = *(const uint4*)(p0 + 8);
        b1.q[0] = *(const uint4*)(p1); b1.q[1] = *(const uint4*)(p1 + 8);
        __builtin_prefetch(p0 + 512, 0, 1);         // next K-slice -> global_prefetch_b8
        __builtin_prefetch(p1 + 512, 0, 1);
        acc0 = wmma_bf16(a.v, b0.v, acc0);
        acc1 = wmma_bf16(a.v, b1.v, acc1);
    }

    // C layout: VGPR j -> row = j + 8*hi (within 16-row tile), col = lane&15
    const int row = mt * 16 + hi * 8;
    const int col = lane & 15;
#pragma unroll
    for (int j = 0; j < 8; ++j) {
        zs[g][row + j][col]      = acc0[j];
        zs[g][row + j][16 + col] = acc1[j];
    }
    __syncthreads();

    // Pointwise LSTM update: 32 batch x 32 cols = 1024 elems over 256 threads.
    for (int p = tid; p < BATCH * 32; p += 256) {
        int bb = p >> 5, jj = p & 31;
        int j  = n0 + jj;
        float zi = zs[0][bb][jj] + bh[j];
        float zf = zs[1][bb][jj] + bh[HIDDEN + j];
        float zg = zs[2][bb][jj] + bh[2 * HIDDEN + j];
        float zo = zs[3][bb][jj] + bh[3 * HIDDEN + j];
        if (useGather) {                              // x@Wi == Wi[tok,:] (one-hot x)
            int tk = tok[bb];
            zi += wi_gather(WiP, tk, j);
            zf += wi_gather(WiP, tk, HIDDEN + j);
            zg += wi_gather(WiP, tk, 2 * HIDDEN + j);
            zo += wi_gather(WiP, tk, 3 * HIDDEN + j);
        }
        float cOld = c[bb * HIDDEN + j];
        float cNew = sigm(zf) * cOld + sigm(zi) * tanhf(zg);
        c[bb * HIDDEN + j] = cNew;
        float hNew = sigm(zo) * tanhf(cNew);
        hOut[bb * HIDDEN + j] = f2bf(hNew);
    }
}

// -------------------- logits GEMM: h(32x1024) @ Wp(1024x8192) + bp --------------------
// WG = 8 waves, each one 16x16 tile: mt = w>>2, nt16 = w&3; WG covers 64 cols. 128 WGs.
__global__ void logits_kernel(const unsigned short* __restrict__ h,
                              const unsigned short* __restrict__ WpP,
                              const float* __restrict__ bp,
                              float* __restrict__ out, int t) {
    const int tid  = threadIdx.x;
    const int wave = tid >> 5, lane = tid & 31;
    const int mt = wave >> 2, nt16 = wave & 3;
    const int ntG = blockIdx.x * 4 + nt16;
    const int m = lane & 15, hi = lane >> 4;
    const unsigned short* Arow = h + (size_t)(mt * 16 + m) * HIDDEN + hi * 16;
    const int ktCount = HIDDEN >> 5;                 // 32

    f32x8 acc = {};
    for (int kt = 0; kt < ktCount; ++kt) {
        Frag a, b;
        a.q[0] = *(const uint4*)(Arow + kt * 32);
        a.q[1] = *(const uint4*)(Arow + kt * 32 + 8);
        const unsigned short* p = WpP + (((size_t)ntG * ktCount + kt) * 32 + lane) * 16;
        b.q[0] = *(const uint4*)(p); b.q[1] = *(const uint4*)(p + 8);
        __builtin_prefetch(p + 512, 0, 1);
        acc = wmma_bf16(a.v, b.v, acc);
    }

    const int n = ntG * 16 + (lane & 15);
    const float bias = bp[n];
    const int rbase = mt * 16 + hi * 8;
#pragma unroll
    for (int j = 0; j < 8; ++j) {
        int bb = rbase + j;
        out[((size_t)bb * TSTEPS + t) * VOCAB + n] = acc[j] + bias;
    }
}

// -------------------- Gumbel-argmax sampling + one-hot pred --------------------
__global__ void sample_kernel(const float* __restrict__ logits, float* __restrict__ preds,
                              int* __restrict__ tok, int t) {
    __shared__ float rv[256];
    __shared__ int   ri[256];
    __shared__ int   sBest;
    const int b = blockIdx.x, tid = threadIdx.x;
    const float* row = logits + ((size_t)b * TSTEPS + t) * VOCAB;

    float best = -3.4e38f; int bidx = 0;
    unsigned seedBase = (unsigned)(t * BATCH + b) * 0x9E3779B9u + 0x85EBCA6Bu;
    for (int v = tid; v < VOCAB; v += 256) {
        unsigned x = seedBase + (unsigned)v * 0x2545F491u;
        x = x * 747796405u + 2891336453u;
        unsigned w = ((x >> ((x >> 28) + 4u)) ^ x) * 277803737u;
        w = (w >> 22) ^ w;
        float u = fmaxf((float)(w >> 8) * (1.0f / 16777216.0f), 1e-7f);
        float gum = -__logf(-__logf(u));
        float s = row[v] + gum;
        if (s > best) { best = s; bidx = v; }
    }
    rv[tid] = best; ri[tid] = bidx;
    __syncthreads();
    for (int s = 128; s > 0; s >>= 1) {
        if (tid < s) {
            float ov = rv[tid + s]; int oi = ri[tid + s];
            if (ov > rv[tid] || (ov == rv[tid] && oi < ri[tid])) { rv[tid] = ov; ri[tid] = oi; }
        }
        __syncthreads();
    }
    if (tid == 0) { sBest = ri[0]; tok[b] = ri[0]; }
    __syncthreads();
    float* prow = preds + ((size_t)b * TSTEPS + t) * VOCAB;
    for (int v = tid; v < VOCAB; v += 256)
        prow[v] = (v == sBest) ? 1.0f : 0.0f;
}

// -------------------- launch --------------------
extern "C" void kernel_launch(void* const* d_in, const int* in_sizes, int n_in,
                              void* d_out, int out_size, void* d_ws, size_t ws_size,
                              hipStream_t stream) {
    (void)in_sizes; (void)n_in; (void)out_size; (void)ws_size;
    const float* inputs = (const float*)d_in[0];
    const float* Wi     = (const float*)d_in[1];
    const float* Wh     = (const float*)d_in[2];
    const float* bh     = (const float*)d_in[3];
    const float* Wp     = (const float*)d_in[4];
    const float* bp     = (const float*)d_in[5];
    const float* c0     = (const float*)d_in[6];
    // d_in[7] = h0 (zeros; h-contribution at t=0 is exactly zero, so unused)

    char* ws = (char*)d_ws;
    unsigned short* WiP = (unsigned short*)ws;  ws += (size_t)VOCAB  * GATES * 2;  // 64 MB
    unsigned short* WhP = (unsigned short*)ws;  ws += (size_t)HIDDEN * GATES * 2;  //  8 MB
    unsigned short* WpP = (unsigned short*)ws;  ws += (size_t)HIDDEN * VOCAB * 2;  // 16 MB
    unsigned short* x0  = (unsigned short*)ws;  ws += (size_t)BATCH  * VOCAB * 2;
    unsigned short* hbf = (unsigned short*)ws;  ws += (size_t)BATCH  * HIDDEN * 2;
    float* cbuf         = (float*)ws;           ws += (size_t)BATCH  * HIDDEN * 4;
    int*   tok          = (int*)ws;

    float* logits = (float*)d_out;
    float* preds  = (float*)d_out + (size_t)BATCH * TSTEPS * VOCAB;

    pack_weight_kernel<<<(unsigned)(((size_t)VOCAB * GATES + 255) / 256), 256, 0, stream>>>(Wi, WiP, VOCAB, 12);
    pack_weight_kernel<<<(unsigned)(((size_t)HIDDEN * GATES + 255) / 256), 256, 0, stream>>>(Wh, WhP, HIDDEN, 12);
    pack_weight_kernel<<<(unsigned)(((size_t)HIDDEN * VOCAB + 255) / 256), 256, 0, stream>>>(Wp, WpP, HIDDEN, 13);
    cvt_x0_kernel<<<(BATCH * VOCAB) / 256, 256, 0, stream>>>(inputs, x0);
    init_c_kernel<<<(BATCH * HIDDEN + 255) / 256, 256, 0, stream>>>(c0, cbuf);

    for (int t = 0; t < TSTEPS; ++t) {
        if (t == 0)
            lstm_step_kernel<<<HIDDEN / 32, 256, 0, stream>>>(x0, VOCAB, WiP, VOCAB >> 5,
                                                              WiP, bh, tok, cbuf, hbf, 0);
        else
            lstm_step_kernel<<<HIDDEN / 32, 256, 0, stream>>>(hbf, HIDDEN, WhP, HIDDEN >> 5,
                                                              WiP, bh, tok, cbuf, hbf, 1);
        logits_kernel<<<VOCAB / 64, 256, 0, stream>>>(hbf, WpP, bp, logits, t);
        sample_kernel<<<BATCH, 256, 0, stream>>>(logits, preds, tok, t);
    }
}
